// SelfMultiHeadAttention_13013750907270
// MI455X (gfx1250) — matrix operand order repaired and checked by reading, not compiled, not run on hardware
//
#include <hip/hip_runtime.h>
#include <hip/hip_bf16.h>

// ---------- types ----------
typedef __bf16 bhalf;
typedef __bf16 v16bf __attribute__((ext_vector_type(16)));
typedef float  v8f   __attribute__((ext_vector_type(8)));
typedef unsigned int u32x4 __attribute__((ext_vector_type(4)));

union Frag16 { v16bf v; u32x4 q[2]; };

#define B_DIM   2
#define S_LEN   2048
#define F_DIM   1024
#define H_DIM   16
#define D_DIM   64
#define HD_DIM  1024
#define NQKV    3072
#define M_ROWS  4096   // B*S

// ---------- helpers ----------
__device__ __forceinline__ bhalf f2bf(float f) {
  unsigned u = __builtin_bit_cast(unsigned, f);
  unsigned r = u + 0x7FFFu + ((u >> 16) & 1u);   // round-to-nearest-even
  unsigned short h = (unsigned short)(r >> 16);
  return __builtin_bit_cast(bhalf, h);
}

__device__ __forceinline__ v8f zero8() {
  v8f z = {0.f, 0.f, 0.f, 0.f, 0.f, 0.f, 0.f, 0.f};
  return z;
}

__device__ __forceinline__ v8f wmma_bf16(v16bf a, v16bf b, v8f c) {
  // D = A(16x32 bf16) * B(32x16 bf16) + C(16x16 f32)
  return __builtin_amdgcn_wmma_f32_16x16x32_bf16(
      /*neg_a=*/false, a, /*neg_b=*/false, b,
      /*c_mod=*/(short)0, c, /*reuse_a=*/false, /*reuse_b=*/false);
}

// Async DMA: per-lane 16B global -> LDS, tracked by ASYNCcnt (one instr/wave).
__device__ __forceinline__ void async_copy_b128(const bhalf* gsrc, bhalf* ldst) {
  unsigned lds_addr = (unsigned)(unsigned long long)(const void*)ldst; // AS3 offset = addr[31:0]
  unsigned long long gaddr = (unsigned long long)(const void*)gsrc;
  asm volatile("global_load_async_to_lds_b128 %0, %1, off"
               :: "v"(lds_addr), "v"(gaddr)
               : "memory");
}
#define WAIT_ASYNCCNT(n) asm volatile("s_wait_asynccnt " #n ::: "memory")

// A-matrix 16x32 bf16 fragment (ISA 7.12.2): lane L holds row m0+(L&15);
// elems 0..7 -> K = k0 + (L>=16?8:0) + i ; elems 8..15 -> +16 more.
__device__ __forceinline__ v16bf load_a_frag(const bhalf* base, int ld, int m0,
                                             int k0, int lane) {
  const bhalf* p = base + (size_t)(m0 + (lane & 15)) * ld + k0 + ((lane >> 4) << 3);
  Frag16 f;
  f.q[0] = *(const u32x4*)p;
  f.q[1] = *(const u32x4*)(p + 16);
  return f.v;
}

// B-matrix 32x16 bf16 fragment: lane L holds column n0+(L&15);
// elems 0..7 -> K = k0 + (L>=16?8:0) + i ; elems 8..15 -> +16 more (matches A).
// Source must be column-major-in-K: row n of `base` holds that column's K run.
__device__ __forceinline__ v16bf load_b_frag(const bhalf* base, int ld, int n0,
                                             int k0, int lane) {
  const bhalf* p = base + (size_t)(n0 + (lane & 15)) * ld + k0 + ((lane >> 4) << 3);
  Frag16 f;
  f.q[0] = *(const u32x4*)p;
  f.q[1] = *(const u32x4*)(p + 16);
  return f.v;
}

// A-fragment from a wave-private 16x32 bf16 LDS tile (row stride 32)
__device__ __forceinline__ v16bf load_a_frag_lds(const bhalf* base, int lane) {
  const bhalf* p = base + (lane & 15) * 32 + ((lane >> 4) << 3);
  Frag16 f;
  f.q[0] = *(const u32x4*)p;
  f.q[1] = *(const u32x4*)(p + 16);
  return f.v;
}

// ---------- kernels ----------
__global__ void k_f32_to_bf16(const float* __restrict__ in,
                              bhalf* __restrict__ out, int n) {
  int i = blockIdx.x * blockDim.x + threadIdx.x;
  if (i < n) out[i] = f2bf(in[i]);
}

// QKV GEMM: C[M=4096, N=3072] = Xbf[4096,1024] * Wqkv[3072,1024]^T
// wave tile 32x64, block (8 waves) tile 64x256.
// Epilogue scatters to Q[bh][s][d], K[bh][s][d], V^T[bh][d][s] (bf16).
__global__ void k_gemm_qkv(const bhalf* __restrict__ xb,
                           const bhalf* __restrict__ wb,
                           bhalf* __restrict__ qbuf,
                           bhalf* __restrict__ kbuf,
                           bhalf* __restrict__ vbuf) {
  const int lane = threadIdx.x & 31;
  const int wave = threadIdx.x >> 5;
  const int m0 = blockIdx.x * 64 + (wave >> 2) * 32;
  const int n0 = blockIdx.y * 256 + (wave & 3) * 64;

  v8f acc[2][4];
#pragma unroll
  for (int i = 0; i < 2; ++i)
#pragma unroll
    for (int j = 0; j < 4; ++j) acc[i][j] = zero8();

  for (int kk = 0; kk < F_DIM; kk += 32) {
    if (kk + 32 < F_DIM) {
      __builtin_prefetch(xb + (size_t)(m0 + (lane & 15)) * F_DIM + kk + 32, 0, 1);
      __builtin_prefetch(wb + (size_t)(n0 + (lane & 15)) * F_DIM + kk + 32, 0, 1);
    }
    v16bf a0 = load_a_frag(xb, F_DIM, m0,      kk, lane);
    v16bf a1 = load_a_frag(xb, F_DIM, m0 + 16, kk, lane);
    v16bf b0 = load_b_frag(wb, F_DIM, n0,      kk, lane);
    v16bf b1 = load_b_frag(wb, F_DIM, n0 + 16, kk, lane);
    v16bf b2 = load_b_frag(wb, F_DIM, n0 + 32, kk, lane);
    v16bf b3 = load_b_frag(wb, F_DIM, n0 + 48, kk, lane);
    acc[0][0] = wmma_bf16(a0, b0, acc[0][0]);
    acc[0][1] = wmma_bf16(a0, b1, acc[0][1]);
    acc[0][2] = wmma_bf16(a0, b2, acc[0][2]);
    acc[0][3] = wmma_bf16(a0, b3, acc[0][3]);
    acc[1][0] = wmma_bf16(a1, b0, acc[1][0]);
    acc[1][1] = wmma_bf16(a1, b1, acc[1][1]);
    acc[1][2] = wmma_bf16(a1, b2, acc[1][2]);
    acc[1][3] = wmma_bf16(a1, b3, acc[1][3]);
  }

  // C/D layout: VGPR r in lane l -> row r + (l>=16 ? 8 : 0), col (l&15)
#pragma unroll
  for (int mi = 0; mi < 2; ++mi)
#pragma unroll
    for (int ni = 0; ni < 4; ++ni)
#pragma unroll
      for (int r = 0; r < 8; ++r) {
        int M = m0 + mi * 16 + r + ((lane >> 4) << 3);
        int N = n0 + ni * 16 + (lane & 15);
        bhalf v = f2bf(acc[mi][ni][r]);
        int b = M >> 11;             // /S_LEN
        int s = M & (S_LEN - 1);
        int which = N >> 10;         // 0=q 1=k 2=v
        int h = (N >> 6) & 15;
        int d = N & 63;
        int bh = b * H_DIM + h;
        if (which == 0)
          qbuf[((size_t)bh * S_LEN + s) * D_DIM + d] = v;
        else if (which == 1)
          kbuf[((size_t)bh * S_LEN + s) * D_DIM + d] = v;
        else
          vbuf[((size_t)bh * D_DIM + d) * S_LEN + s] = v;  // transposed
      }
}

// Flash attention, cooperative async-LDS staging of K/V chunks.
// Block = 8 waves = 128 q rows of one (b,h). Keys streamed 32 at a time:
//   K chunk  32x64 bf16 (4KB) + V^T chunk 64x32 bf16 (4KB), double buffered.
#define KV_K_OFF 0
#define KV_V_OFF 2048   // elements
#define KV_ELEMS 4096   // per buffer

__device__ __forceinline__ void stage_kv(const bhalf* __restrict__ K,
                                         const bhalf* __restrict__ VT,
                                         bhalf* buf, int kc, int tid) {
  // K rows kc..kc+31 are contiguous 4KB in [S,D] layout: linear 16B per thread
  async_copy_b128(K + (size_t)kc * D_DIM + tid * 8, buf + KV_K_OFF + tid * 8);
  // V^T rows d=0..63, 32 keys (64B) each, stride S_LEN
  int d = tid >> 2, seg = tid & 3;
  async_copy_b128(VT + (size_t)d * S_LEN + kc + seg * 8,
                  buf + KV_V_OFF + d * 32 + seg * 8);
}

__global__ void k_attn(const bhalf* __restrict__ qbuf,
                       const bhalf* __restrict__ kbuf,
                       const bhalf* __restrict__ vbuf,
                       bhalf* __restrict__ obuf) {
  __shared__ bhalf kv_lds[2][KV_ELEMS];   // 16 KB double-buffered K/V chunks
  __shared__ bhalf plds[8][16 * 32];      // 8 KB wave-private P tiles

  const int tid  = threadIdx.x;
  const int lane = tid & 31;
  const int wave = tid >> 5;
  const int bh = blockIdx.y;                      // 0..31
  const int q0 = blockIdx.x * 128 + wave * 16;    // q-row tile base

  const bhalf* Q  = qbuf + (size_t)bh * S_LEN * D_DIM;
  const bhalf* K  = kbuf + (size_t)bh * S_LEN * D_DIM;
  const bhalf* VT = vbuf + (size_t)bh * D_DIM * S_LEN;

  // Q fragments held for the whole loop (D=64 -> two K=32 chunks)
  v16bf qa0 = load_a_frag(Q, D_DIM, q0, 0,  lane);
  v16bf qa1 = load_a_frag(Q, D_DIM, q0, 32, lane);

  v8f o[4];
#pragma unroll
  for (int t = 0; t < 4; ++t) o[t] = zero8();
  float mrow[8], lrow[8];
#pragma unroll
  for (int r = 0; r < 8; ++r) { mrow[r] = -1e30f; lrow[r] = 0.f; }

  const float scale = 0.125f;  // 1/sqrt(64)

  stage_kv(K, VT, &kv_lds[0][0], 0, tid);   // prime buffer 0

  for (int kc = 0; kc < S_LEN; kc += 32) {
    const int cur = (kc >> 5) & 1;
    if (kc + 32 < S_LEN) {
      stage_kv(K, VT, &kv_lds[cur ^ 1][0], kc + 32, tid);  // prefetch next
      WAIT_ASYNCCNT(2);   // in-order: oldest 2 (current buffer) complete
    } else {
      WAIT_ASYNCCNT(0);
    }
    __syncthreads();      // all waves' staging of `cur` visible in LDS

    const bhalf* bk = &kv_lds[cur][KV_K_OFF];
    const bhalf* bv = &kv_lds[cur][KV_V_OFF];

    // scores: two 16x16 tiles covering 32 keys, K-dim = D = 64 (chained WMMA)
    v8f s0 = wmma_bf16(qa0, load_b_frag(bk, D_DIM, 0,  0,  lane), zero8());
    s0     = wmma_bf16(qa1, load_b_frag(bk, D_DIM, 0,  32, lane), s0);
    v8f s1 = wmma_bf16(qa0, load_b_frag(bk, D_DIM, 16, 0,  lane), zero8());
    s1     = wmma_bf16(qa1, load_b_frag(bk, D_DIM, 16, 32, lane), s1);

    // online softmax, per accumulator row r (rows live across 16-lane halves)
#pragma unroll
    for (int r = 0; r < 8; ++r) {
      float a = s0[r] * scale;
      float b = s1[r] * scale;
      float mx = fmaxf(a, b);
#pragma unroll
      for (int off = 8; off >= 1; off >>= 1)
        mx = fmaxf(mx, __shfl_xor(mx, off, 32));
      float mn = fmaxf(mrow[r], mx);
      float alpha = __expf(mrow[r] - mn);
      float e0 = __expf(a - mn);
      float e1 = __expf(b - mn);
      float sum = e0 + e1;
#pragma unroll
      for (int off = 8; off >= 1; off >>= 1)
        sum += __shfl_xor(sum, off, 32);
      lrow[r] = lrow[r] * alpha + sum;
      mrow[r] = mn;
#pragma unroll
      for (int t = 0; t < 4; ++t) o[t][r] *= alpha;
      // stash P (C-layout) into LDS as bf16 for A-layout reload (wave-local)
      int row = r + ((lane >> 4) << 3);
      plds[wave][row * 32 + (lane & 15)]      = f2bf(e0);
      plds[wave][row * 32 + 16 + (lane & 15)] = f2bf(e1);
    }

    // P(16x32) * V(32x64): 4 N-tiles, V from LDS (ld = 32 within chunk)
    v16bf pa = load_a_frag_lds(&plds[wave][0], lane);
#pragma unroll
    for (int t = 0; t < 4; ++t)
      o[t] = wmma_bf16(pa, load_b_frag(bv, 32, t * 16, 0, lane), o[t]);

    __syncthreads();      // don't let any wave overwrite `cur^1` early
  }

  // normalize + store to [b, s, h*64+d] bf16
  const int b = bh >> 4;
  const int h = bh & 15;
#pragma unroll
  for (int t = 0; t < 4; ++t)
#pragma unroll
    for (int r = 0; r < 8; ++r) {
      int row = q0 + r + ((lane >> 4) << 3);
      int d = t * 16 + (lane & 15);
      float v = o[t][r] / lrow[r];
      obuf[((size_t)b * S_LEN + row) * HD_DIM + h * D_DIM + d] = f2bf(v);
    }
}

// Output projection: out[M=4096, F=1024] = O[4096,1024] * Wproj[1024,1024]^T + b
__global__ void k_gemm_proj(const bhalf* __restrict__ ob,
                            const bhalf* __restrict__ wb,
                            const float* __restrict__ bias,
                            float* __restrict__ out) {
  const int lane = threadIdx.x & 31;
  const int wave = threadIdx.x >> 5;
  const int m0 = blockIdx.x * 64 + (wave >> 2) * 32;
  const int n0 = blockIdx.y * 256 + (wave & 3) * 64;

  v8f acc[2][4];
#pragma unroll
  for (int i = 0; i < 2; ++i)
#pragma unroll
    for (int j = 0; j < 4; ++j) acc[i][j] = zero8();

  for (int kk = 0; kk < HD_DIM; kk += 32) {
    if (kk + 32 < HD_DIM) {
      __builtin_prefetch(ob + (size_t)(m0 + (lane & 15)) * HD_DIM + kk + 32, 0, 1);
      __builtin_prefetch(wb + (size_t)(n0 + (lane & 15)) * HD_DIM + kk + 32, 0, 1);
    }
    v16bf a0 = load_a_frag(ob, HD_DIM, m0,      kk, lane);
    v16bf a1 = load_a_frag(ob, HD_DIM, m0 + 16, kk, lane);
    v16bf b0 = load_b_frag(wb, HD_DIM, n0,      kk, lane);
    v16bf b1 = load_b_frag(wb, HD_DIM, n0 + 16, kk, lane);
    v16bf b2 = load_b_frag(wb, HD_DIM, n0 + 32, kk, lane);
    v16bf b3 = load_b_frag(wb, HD_DIM, n0 + 48, kk, lane);
    acc[0][0] = wmma_bf16(a0, b0, acc[0][0]);
    acc[0][1] = wmma_bf16(a0, b1, acc[0][1]);
    acc[0][2] = wmma_bf16(a0, b2, acc[0][2]);
    acc[0][3] = wmma_bf16(a0, b3, acc[0][3]);
    acc[1][0] = wmma_bf16(a1, b0, acc[1][0]);
    acc[1][1] = wmma_bf16(a1, b1, acc[1][1]);
    acc[1][2] = wmma_bf16(a1, b2, acc[1][2]);
    acc[1][3] = wmma_bf16(a1, b3, acc[1][3]);
  }

#pragma unroll
  for (int mi = 0; mi < 2; ++mi)
#pragma unroll
    for (int ni = 0; ni < 4; ++ni)
#pragma unroll
      for (int r = 0; r < 8; ++r) {
        int M = m0 + mi * 16 + r + ((lane >> 4) << 3);
        int N = n0 + ni * 16 + (lane & 15);
        out[(size_t)M * HD_DIM + N] = acc[mi][ni][r] + bias[N];
      }
}

// ---------- launch ----------
extern "C" void kernel_launch(void* const* d_in, const int* in_sizes, int n_in,
                              void* d_out, int out_size, void* d_ws, size_t ws_size,
                              hipStream_t stream) {
  const float* x      = (const float*)d_in[0];  // [2,2048,1024]
  const float* w_qkv  = (const float*)d_in[1];  // [3072,1024]
  const float* w_proj = (const float*)d_in[2];  // [1024,1024]
  const float* b_proj = (const float*)d_in[3];  // [1024]
  float* out = (float*)d_out;                   // [2,2048,1024]

  char* ws = (char*)d_ws;
  bhalf* xb  = (bhalf*)ws; ws += (size_t)M_ROWS * F_DIM  * 2;   // 8 MB
  bhalf* wqb = (bhalf*)ws; ws += (size_t)NQKV   * F_DIM  * 2;   // 6 MB
  bhalf* wpb = (bhalf*)ws; ws += (size_t)F_DIM  * HD_DIM * 2;   // 2 MB
  bhalf* qb  = (bhalf*)ws; ws += (size_t)B_DIM * H_DIM * S_LEN * D_DIM * 2; // 8 MB
  bhalf* kb  = (bhalf*)ws; ws += (size_t)B_DIM * H_DIM * S_LEN * D_DIM * 2; // 8 MB
  bhalf* vb  = (bhalf*)ws; ws += (size_t)B_DIM * H_DIM * S_LEN * D_DIM * 2; // 8 MB
  bhalf* obf = (bhalf*)ws; ws += (size_t)M_ROWS * HD_DIM * 2;   // 8 MB

  const int nx = M_ROWS * F_DIM;     // 4194304
  const int nq = NQKV * F_DIM;       // 3145728
  const int np = F_DIM * HD_DIM;     // 1048576
  k_f32_to_bf16<<<(nx + 255) / 256, 256, 0, stream>>>(x, xb, nx);
  k_f32_to_bf16<<<(nq + 255) / 256, 256, 0, stream>>>(w_qkv, wqb, nq);
  k_f32_to_bf16<<<(np + 255) / 256, 256, 0, stream>>>(w_proj, wpb, np);

  k_gemm_qkv<<<dim3(M_ROWS / 64, NQKV / 256), 256, 0, stream>>>(xb, wqb, qb, kb, vb);
  k_attn<<<dim3(S_LEN / 128, B_DIM * H_DIM), 256, 0, stream>>>(qb, kb, vb, obf);
  k_gemm_proj<<<dim3(M_ROWS / 64, HD_DIM / 256), 256, 0, stream>>>(obf, wpb, b_proj, out);
}
